// SingleHeadGAT_46488726012187
// MI455X (gfx1250) — compile-verified, matrix operand order checked by reading, not run on hardware
//
#include <hip/hip_runtime.h>
#include <hip/hip_bf16.h>

typedef __attribute__((ext_vector_type(16))) __bf16 v16bf;
typedef __attribute__((ext_vector_type(8)))  float  v8f;
typedef __attribute__((ext_vector_type(4)))  float  f32x4;
typedef __attribute__((ext_vector_type(4)))  unsigned int u32x4;

#define HC   256   // H*C
#define CH   64    // channels per head
#define NH   4     // heads
#define NEG_SLOPE 0.2f

union FragBF { v16bf v; u32x4 q[2]; };

// ---------------------------------------------------------------------------
// WMMA GEMM: C[M,256] = A[M,K] @ B[K,256]; fp32 in/out, bf16 WMMA, f32 accum.
// Block = 256 threads = 8 waves arranged 4(M) x 2(N); each wave owns a 16x32
// strip (two 16x16 WMMA tiles sharing one A fragment) -> 64x64 block tile.
// K templated (128/256) and fully unrolled: per k-step per wave =
// 4x global_load_b128 (A) + 4x ds_load_b128 (B) + cvt + 2x v_wmma.
// ---------------------------------------------------------------------------
template <int K>
__global__ __launch_bounds__(256) void gemm_wmma_bf16(
    const float* __restrict__ A, const float* __restrict__ B,
    float* __restrict__ C, int M)
{
    constexpr int Kp = K + 8;                 // pad: kills LDS bank conflicts
    __shared__ __bf16 Bs[64 * Kp];            // column-major 64-col K-panel

    const int nBlk = blockIdx.y * 64;
    for (int idx = threadIdx.x; idx < K * 64; idx += 256) {
        int k = idx >> 6, c = idx & 63;       // coalesced global read
        Bs[c * Kp + k] = (__bf16)B[(size_t)k * HC + nBlk + c];
    }
    __syncthreads();

    const int lane  = threadIdx.x & 31;
    const int wave  = threadIdx.x >> 5;
    const int wm    = wave >> 1;              // 0..3
    const int wn    = wave & 1;               // 0..1
    const int mBase = blockIdx.x * 64 + wm * 16;
    const int nBase = nBlk + wn * 32;
    const int mRow  = mBase + (lane & 15);    // A layout: M = lane&15
    const int kb    = (lane >> 4) << 3;       // lanes 16-31: K offset +8
    // clamp OOB rows to row 0: garbage only lands in D rows that are not stored
    const float* ap = A + (size_t)(mRow < M ? mRow : 0) * K;
    const __bf16* bs0 = Bs + (wn * 32 + (lane & 15)) * Kp;        // tile 0 col
    const __bf16* bs1 = Bs + (wn * 32 + 16 + (lane & 15)) * Kp;   // tile 1 col

    v8f acc0 = {}, acc1 = {};
#pragma unroll
    for (int k0 = 0; k0 < K; k0 += 32) {
        if (k0 + 32 < K) __builtin_prefetch(ap + k0 + 32, 0, 0);

        // A fragment: 16 fp32 -> bf16 (K = k0+kb+0..7 and k0+16+kb+0..7)
        f32x4 a0 = *(const f32x4*)(ap + k0 + kb);
        f32x4 a1 = *(const f32x4*)(ap + k0 + kb + 4);
        f32x4 a2 = *(const f32x4*)(ap + k0 + 16 + kb);
        f32x4 a3 = *(const f32x4*)(ap + k0 + 16 + kb + 4);
        v16bf a;
#pragma unroll
        for (int i = 0; i < 4; ++i) {
            a[i]      = (__bf16)a0[i];
            a[4 + i]  = (__bf16)a1[i];
            a[8 + i]  = (__bf16)a2[i];
            a[12 + i] = (__bf16)a3[i];
        }

        // B fragments: contiguous bf16 in LDS -> two 16B chunks each
        FragBF b0, b1;
        b0.q[0] = *(const u32x4*)(bs0 + k0 + kb);
        b0.q[1] = *(const u32x4*)(bs0 + k0 + 16 + kb);
        b1.q[0] = *(const u32x4*)(bs1 + k0 + kb);
        b1.q[1] = *(const u32x4*)(bs1 + k0 + 16 + kb);

        acc0 = __builtin_amdgcn_wmma_f32_16x16x32_bf16(
                   false, a, false, b0.v, (short)0, acc0, false, false);
        acc1 = __builtin_amdgcn_wmma_f32_16x16x32_bf16(
                   false, a, false, b1.v, (short)0, acc1, false, false);
    }

    // C/D layout: VGPR r -> M = r + 8*(lane>=16), N = lane&15
#pragma unroll
    for (int r = 0; r < 8; ++r) {
        int row = mBase + r + ((lane >> 4) << 3);
        if (row < M) {
            float* cp = C + (size_t)row * HC + nBase + (lane & 15);
            cp[0]  = acc0[r];
            cp[16] = acc1[r];
        }
    }
}

// ---------------------------------------------------------------------------
// Per-node attention scores: a_src[n,h] = sum_c h[n,h,c]*att_src[h,c] (+dst)
// ---------------------------------------------------------------------------
__global__ __launch_bounds__(256) void attn_scores(
    const float* __restrict__ h, const float* __restrict__ attS,
    const float* __restrict__ attD, float* __restrict__ aS,
    float* __restrict__ aD, int N)
{
    __shared__ float ss[256], sd[256];
    int n = blockIdx.x;
    int t = threadIdx.x;
    float v = h[(size_t)n * HC + t];
    ss[t] = v * attS[t];
    sd[t] = v * attD[t];
    __syncthreads();
#pragma unroll
    for (int off = 32; off > 0; off >>= 1) {
        if ((t & 63) < off) { ss[t] += ss[t + off]; sd[t] += sd[t + off]; }
        __syncthreads();
    }
    if ((t & 63) == 0) {
        int hd = t >> 6;
        aS[n * NH + hd] = ss[t];
        aD[n * NH + hd] = sd[t];
    }
}

// monotonic float<->uint key for atomicMax on floats
__device__ __forceinline__ unsigned fkey(float f) {
    unsigned u = __float_as_uint(f);
    return (f >= 0.f) ? (u | 0x80000000u) : ~u;
}
__device__ __forceinline__ float funkey(unsigned k) {
    unsigned u = (k & 0x80000000u) ? (k & 0x7FFFFFFFu) : ~k;
    return __uint_as_float(u);
}

// e = leaky_relu(a_src[src]+a_dst[dst]); segment max into mxu[dst]
__global__ __launch_bounds__(256) void edge_logits(
    const int* __restrict__ ei, int E, int N,
    const float* __restrict__ aS, const float* __restrict__ aD,
    float* __restrict__ ebuf, unsigned* __restrict__ mxu)
{
    int g = blockIdx.x * 256 + threadIdx.x;
    int total = (E + N) * NH;
    if (g >= total) return;
    int e = g >> 2, hd = g & 3;
    int s, d;
    if (e < E) { s = ei[e]; d = ei[E + e]; } else { s = d = e - E; }
    float v = aS[s * NH + hd] + aD[d * NH + hd];
    v = v > 0.f ? v : NEG_SLOPE * v;
    ebuf[g] = v;
    atomicMax(&mxu[d * NH + hd], fkey(v));
}

// p = exp(e - max[dst]); segment sum into denom[dst]; ebuf <- p
__global__ __launch_bounds__(256) void edge_softmax(
    const int* __restrict__ ei, int E, int N,
    float* __restrict__ ebuf, const unsigned* __restrict__ mxu,
    float* __restrict__ denom)
{
    int g = blockIdx.x * 256 + threadIdx.x;
    int total = (E + N) * NH;
    if (g >= total) return;
    int e = g >> 2, hd = g & 3;
    int d = (e < E) ? ei[E + e] : (e - E);
    float m = funkey(mxu[d * NH + hd]);
    float p = __expf(ebuf[g] - m);
    ebuf[g] = p;
    atomicAdd(&denom[d * NH + hd], p);
}

// agg[dst] += h[src] * (p / denom[dst]); one block per edge, 256 feats
__global__ __launch_bounds__(256) void edge_msg(
    const int* __restrict__ ei, int E, int N,
    const float* __restrict__ h, const float* __restrict__ pbuf,
    const float* __restrict__ denom, float* __restrict__ agg)
{
    int e = blockIdx.x;
    int f = threadIdx.x;
    int s, d;
    if (e < E) { s = ei[e]; d = ei[E + e]; } else { s = d = e - E; }
    int hd = f >> 6;
    float alpha = pbuf[e * NH + hd] / denom[d * NH + hd];
    atomicAdd(&agg[(size_t)d * HC + f], h[(size_t)s * HC + f] * alpha);
}

// in-place: p[i] = elu(p[i] + bias[i & 255])
__global__ __launch_bounds__(256) void bias_elu(
    float* __restrict__ p, const float* __restrict__ b, long long n)
{
    long long i = (long long)blockIdx.x * 256 + threadIdx.x;
    if (i >= n) return;
    float v = p[i] + b[i & (HC - 1)];
    p[i] = v > 0.f ? v : (__expf(v) - 1.f);
}

// layer-2 finalize: mean over heads + b2 + elu, then pooled atomic accumulate
__global__ __launch_bounds__(256) void head_mean_pool(
    const float* __restrict__ agg2, const float* __restrict__ b2,
    const int* __restrict__ batch, float* __restrict__ pooled,
    float* __restrict__ cnt, int N)
{
    int g = blockIdx.x * 256 + threadIdx.x;
    if (g >= N * CH) return;
    int n = g >> 6, c = g & 63;
    const float* row = agg2 + (size_t)n * HC;
    float v = 0.25f * (row[c] + row[CH + c] + row[2 * CH + c] + row[3 * CH + c]) + b2[c];
    v = v > 0.f ? v : (__expf(v) - 1.f);
    int bg = batch[n];
    atomicAdd(&pooled[bg * CH + c], v);
    if (c == 0) atomicAdd(&cnt[bg], 1.f);
}

// out[g,a] = (pooled[g]/max(cnt,1)) . Wa[:,a] + ba[a]
__global__ __launch_bounds__(256) void final_mlp(
    const float* __restrict__ pooled, const float* __restrict__ cnt,
    const float* __restrict__ Wa, const float* __restrict__ ba,
    float* __restrict__ out, int A, int total)
{
    int idx = blockIdx.x * 256 + threadIdx.x;
    if (idx >= total) return;
    int g = idx / A, a = idx % A;
    float inv = 1.f / fmaxf(cnt[g], 1.f);
    float acc = ba[a];
#pragma unroll 8
    for (int c = 0; c < CH; ++c)
        acc += pooled[g * CH + c] * inv * Wa[c * A + a];
    out[idx] = acc;
}

__global__ __launch_bounds__(256) void zero_fill(float* p, long long n)
{
    long long i = (long long)blockIdx.x * 256 + threadIdx.x;
    if (i < n) p[i] = 0.f;
}

// ---------------------------------------------------------------------------
extern "C" void kernel_launch(void* const* d_in, const int* in_sizes, int n_in,
                              void* d_out, int out_size, void* d_ws, size_t ws_size,
                              hipStream_t stream)
{
    const float* x     = (const float*)d_in[0];
    const int*   ei    = (const int*)  d_in[1];
    const int*   batch = (const int*)  d_in[2];
    const float* W1    = (const float*)d_in[5];
    const float* attS1 = (const float*)d_in[6];
    const float* attD1 = (const float*)d_in[7];
    const float* b1    = (const float*)d_in[8];
    const float* W2    = (const float*)d_in[9];
    const float* attS2 = (const float*)d_in[10];
    const float* attD2 = (const float*)d_in[11];
    const float* b2    = (const float*)d_in[12];
    const float* Wa    = (const float*)d_in[13];
    const float* ba    = (const float*)d_in[14];
    float* out = (float*)d_out;

    const int N   = in_sizes[2];
    const int Fin = in_sizes[0] / N;
    const int E   = in_sizes[1] / 2;
    const int E2  = E + N;
    const int A   = in_sizes[14];
    const int G   = out_size / A;

    // workspace carve-out
    char*  ws = (char*)d_ws;
    size_t o  = 0;
    auto carve = [&](size_t bytes) -> void* {
        void* p = ws + o;
        o += (bytes + 255) & ~(size_t)255;
        return p;
    };
    float*    bufA   = (float*)   carve((size_t)N * HC * 4);   // projections
    float*    bufB   = (float*)   carve((size_t)N * HC * 4);   // aggregations / h1
    float*    ebuf   = (float*)   carve((size_t)E2 * NH * 4);  // logits -> p
    float*    aS     = (float*)   carve((size_t)N * NH * 4);
    float*    aD     = (float*)   carve((size_t)N * NH * 4);
    unsigned* mxu    = (unsigned*)carve((size_t)N * NH * 4);
    float*    denom  = (float*)   carve((size_t)N * NH * 4);
    float*    pooled = (float*)   carve((size_t)G * CH * 4);
    float*    cnt    = (float*)   carve((size_t)G * 4);

    const dim3 T(256);
    auto zgrid = [](long long n) { return dim3((unsigned)((n + 255) / 256)); };
    const dim3 gGemm((N + 63) / 64, HC / 64);
    const dim3 gEdge(zgrid((long long)E2 * NH));

    auto run_layer_edges = [&](const float* hproj, const float* as_, const float* ad_,
                               float* agg) {
        zero_fill<<<zgrid((long long)N * NH), T, 0, stream>>>((float*)mxu, (long long)N * NH);
        zero_fill<<<zgrid((long long)N * NH), T, 0, stream>>>(denom, (long long)N * NH);
        zero_fill<<<zgrid((long long)N * HC), T, 0, stream>>>(agg, (long long)N * HC);
        edge_logits <<<gEdge, T, 0, stream>>>(ei, E, N, as_, ad_, ebuf, mxu);
        edge_softmax<<<gEdge, T, 0, stream>>>(ei, E, N, ebuf, mxu, denom);
        edge_msg    <<<dim3(E2), T, 0, stream>>>(ei, E, N, hproj, ebuf, denom, agg);
    };

    // ---- Layer 1: h1p = x @ W1 (WMMA), edges, elu(+b1)
    if (Fin == 128)
        gemm_wmma_bf16<128><<<gGemm, T, 0, stream>>>(x, W1, bufA, N);
    else
        gemm_wmma_bf16<256><<<gGemm, T, 0, stream>>>(x, W1, bufA, N);
    attn_scores<<<dim3(N), T, 0, stream>>>(bufA, attS1, attD1, aS, aD, N);
    run_layer_edges(bufA, aS, aD, bufB);
    bias_elu<<<zgrid((long long)N * HC), T, 0, stream>>>(bufB, b1, (long long)N * HC);

    // ---- Layer 2: h2p = h1 @ W2 (WMMA), edges
    gemm_wmma_bf16<256><<<gGemm, T, 0, stream>>>(bufB, W2, bufA, N);
    attn_scores<<<dim3(N), T, 0, stream>>>(bufA, attS2, attD2, aS, aD, N);
    run_layer_edges(bufA, aS, aD, bufB);

    // ---- head-mean + b2 + elu + per-graph pooling, then final MLP
    zero_fill<<<zgrid((long long)G * CH), T, 0, stream>>>(pooled, (long long)G * CH);
    zero_fill<<<zgrid((long long)G),      T, 0, stream>>>(cnt,    (long long)G);
    head_mean_pool<<<zgrid((long long)N * CH), T, 0, stream>>>(bufB, b2, batch, pooled, cnt, N);
    final_mlp<<<zgrid(out_size), T, 0, stream>>>(pooled, cnt, Wa, ba, out, A, out_size);
}
